// DeepResidualGRU_67843303408280
// MI455X (gfx1250) — matrix operand order repaired and checked by reading, not compiled, not run on hardware
//
#include <hip/hip_runtime.h>

#define HID   1024
#define INP   1024
#define OUTD  512
#define SEQ   512
#define BATCH 64
#define NPOOL 256   // HID / 4

typedef __bf16 bf16_t;
typedef __attribute__((ext_vector_type(16))) __bf16 v16bf;
typedef __attribute__((ext_vector_type(8)))  __bf16 v8bf;
typedef __attribute__((ext_vector_type(8)))  float  v8f;

// ---- fragment loaders ------------------------------------------------------
// A (16x32 bf16, MxK): lane l holds row M=l%16; half h=l/16 holds K = k0+8h..+7
// and K = k0+16+8h..+7 (ISA 7.12.2).  Two aligned b128 loads.
__device__ __forceinline__ v16bf load_a_bf(const bf16_t* base, long ldk,
                                           int m, int k0, int half) {
  const bf16_t* p0 = base + (long)m * ldk + k0 + 8 * half;
  v8bf lo = *(const v8bf*)(p0);
  v8bf hi = *(const v8bf*)(p0 + 16);
  return __builtin_shufflevector(lo, hi, 0, 1, 2, 3, 4, 5, 6, 7,
                                 8, 9, 10, 11, 12, 13, 14, 15);
}

// B (32x16 bf16, KxN): lane l holds col N=l%16; half h holds K = k0+8h..+7
// and K = k0+16+8h..+7 (same order as A), along a row of the row-major [N,K] weight.
__device__ __forceinline__ v16bf load_b_bf(const bf16_t* w, long ldk,
                                           int n, int k0, int half) {
  const bf16_t* p0 = w + (long)n * ldk + k0 + 8 * half;
  v8bf lo = *(const v8bf*)(p0);
  v8bf hi = *(const v8bf*)(p0 + 16);
  return __builtin_shufflevector(lo, hi, 0, 1, 2, 3, 4, 5, 6, 7,
                                 8, 9, 10, 11, 12, 13, 14, 15);
}

__device__ __forceinline__ v8f wmma_bf16(v16bf a, v16bf b, v8f c) {
  return __builtin_amdgcn_wmma_f32_16x16x32_bf16(false, a, false, b,
                                                 (short)0, c, false, false);
}

__device__ __forceinline__ float sigm(float x) {
  return 1.0f / (1.0f + __expf(-x));
}

// ---- fused GRU gates + residual lin + BatchNorm(batch stats) + ReLU + MaxPool4
// grid: 64 blocks (16 features each), 128 threads (4 waves, one 16-row M-tile each)
__global__ __launch_bounds__(128, 1)
void gate_bn_pool_kernel(const bf16_t* __restrict__ xb,   // A for gi / lin
                         const bf16_t* __restrict__ hb,   // A for gh
                         const float* __restrict__ hf,    // fp32 h for z*h
                         const bf16_t* __restrict__ wih,
                         const bf16_t* __restrict__ whh,
                         const bf16_t* __restrict__ wlin,
                         const float* __restrict__ bih,
                         const float* __restrict__ bhh,
                         const float* __restrict__ blin,
                         const float* __restrict__ bng,
                         const float* __restrict__ bnb,
                         bf16_t* __restrict__ pooled) {
  __shared__ float2 part[4][2][16];
  __shared__ float  tile[64][16];

  const int tid  = threadIdx.x;
  const int wave = tid >> 5;
  const int lane = tid & 31;
  const int half = lane >> 4;
  const int nfl  = lane & 15;
  const int m0   = wave * 16;
  const int n0   = blockIdx.x * 16;
  const int arow = m0 + nfl;

  const v8f vz = {0.f, 0.f, 0.f, 0.f, 0.f, 0.f, 0.f, 0.f};
  v8f acc[7];
#pragma unroll
  for (int i = 0; i < 7; ++i) acc[i] = vz;

  for (int k0 = 0; k0 < HID; k0 += 32) {
    v16bf ax = load_a_bf(xb, HID, arow, k0, half);
    v16bf ah = load_a_bf(hb, HID, arow, k0, half);
#pragma unroll
    for (int g = 0; g < 3; ++g) {
      v16bf b = load_b_bf(wih, HID, n0 + g * HID + nfl, k0, half);
      acc[g] = wmma_bf16(ax, b, acc[g]);
    }
#pragma unroll
    for (int g = 0; g < 3; ++g) {
      v16bf b = load_b_bf(whh, HID, n0 + g * HID + nfl, k0, half);
      acc[3 + g] = wmma_bf16(ah, b, acc[3 + g]);
    }
    {
      v16bf b = load_b_bf(wlin, HID, n0 + nfl, k0, half);
      acc[6] = wmma_bf16(ax, b, acc[6]);
    }
  }

  const int nf = n0 + nfl;
  const float b_ir = bih[nf], b_iz = bih[HID + nf], b_in = bih[2 * HID + nf];
  const float b_hr = bhh[nf], b_hz = bhh[HID + nf], b_hn = bhh[2 * HID + nf];
  const float b_l = blin[nf], g_ = bng[nf], bb_ = bnb[nf];

  float hpre[8];
  float s = 0.f, sq = 0.f;
#pragma unroll
  for (int r = 0; r < 8; ++r) {
    const int m = m0 + r + 8 * half;                 // C layout: M=r+8*half
    const float rg = sigm(acc[0][r] + b_ir + acc[3][r] + b_hr);
    const float zg = sigm(acc[1][r] + b_iz + acc[4][r] + b_hz);
    const float ng = tanhf(acc[2][r] + b_in + rg * (acc[5][r] + b_hn));
    const float hv = hf[(long)m * HID + nf];
    const float v  = (1.f - zg) * ng + zg * hv + acc[6][r] + b_l;
    hpre[r] = v;
    s += v;
    sq += v * v;
  }
  part[wave][half][nfl] = make_float2(s, sq);
  __syncthreads();

  float ts = 0.f, tq = 0.f;
#pragma unroll
  for (int w = 0; w < 4; ++w) {
    float2 p0 = part[w][0][nfl];
    float2 p1 = part[w][1][nfl];
    ts += p0.x + p1.x;
    tq += p0.y + p1.y;
  }
  const float mean  = ts * (1.f / 64.f);
  const float var   = tq * (1.f / 64.f) - mean * mean;   // biased var
  const float scale = rsqrtf(var + 1e-5f) * g_;
  const float shift = bb_ - mean * scale;

#pragma unroll
  for (int r = 0; r < 8; ++r) {
    const int m = m0 + r + 8 * half;
    const float v = hpre[r] * scale + shift;
    tile[m][nfl] = v > 0.f ? v : 0.f;                   // ReLU
  }
  __syncthreads();

  // MaxPool1d(4) across features, write bf16 [64, 256]
#pragma unroll
  for (int i = 0; i < 2; ++i) {
    const int idx = tid + i * 128;
    const int row = idx >> 2, pc = idx & 3;
    float v = tile[row][pc * 4];
    v = fmaxf(v, tile[row][pc * 4 + 1]);
    v = fmaxf(v, tile[row][pc * 4 + 2]);
    v = fmaxf(v, tile[row][pc * 4 + 3]);
    pooled[(long)row * NPOOL + blockIdx.x * 4 + pc] = (bf16_t)v;
  }
}

// ---- expand: hidden = tanh(pooled @ exp_wT + exp_b)  [64,256]x[256,1024]
// writes the new hidden both as fp32 (gating term) and bf16 (next GEMM A)
__global__ __launch_bounds__(128, 1)
void expand_kernel(const bf16_t* __restrict__ pooled,
                   const bf16_t* __restrict__ wexp,
                   const float* __restrict__ bexp,
                   float* __restrict__ hout,
                   bf16_t* __restrict__ hout_bf) {
  const int tid = threadIdx.x, wave = tid >> 5, lane = tid & 31;
  const int half = lane >> 4, nfl = lane & 15;
  const int m0 = wave * 16, n0 = blockIdx.x * 16;
  const int arow = m0 + nfl;

  v8f acc = {0.f, 0.f, 0.f, 0.f, 0.f, 0.f, 0.f, 0.f};
#pragma unroll
  for (int k0 = 0; k0 < NPOOL; k0 += 32) {
    v16bf a = load_a_bf(pooled, NPOOL, arow, k0, half);
    v16bf b = load_b_bf(wexp, NPOOL, n0 + nfl, k0, half);
    acc = wmma_bf16(a, b, acc);
  }
  const int nf = n0 + nfl;
  const float bv = bexp[nf];
#pragma unroll
  for (int r = 0; r < 8; ++r) {
    const int m = m0 + r + 8 * half;
    const float v = tanhf(acc[r] + bv);
    hout[(long)m * HID + nf]    = v;
    hout_bf[(long)m * HID + nf] = (bf16_t)v;
  }
}

// ---- projection: out[:,t,:] = h @ proj_wT + proj_b  [64,1024]x[1024,512]
__global__ __launch_bounds__(128, 1)
void proj_kernel(const bf16_t* __restrict__ hb,
                 const bf16_t* __restrict__ wproj,
                 const float* __restrict__ bproj,
                 float* __restrict__ out, long orow) {
  const int tid = threadIdx.x, wave = tid >> 5, lane = tid & 31;
  const int half = lane >> 4, nfl = lane & 15;
  const int m0 = wave * 16, n0 = blockIdx.x * 16;
  const int arow = m0 + nfl;

  v8f acc = {0.f, 0.f, 0.f, 0.f, 0.f, 0.f, 0.f, 0.f};
  for (int k0 = 0; k0 < HID; k0 += 32) {
    v16bf a = load_a_bf(hb, HID, arow, k0, half);
    v16bf b = load_b_bf(wproj, HID, n0 + nfl, k0, half);
    acc = wmma_bf16(a, b, acc);
  }
  const int nf = n0 + nfl;
  const float bv = bproj[nf];
#pragma unroll
  for (int r = 0; r < 8; ++r) {
    const int m = m0 + r + 8 * half;
    out[(long)m * orow + nf] = acc[r] + bv;
  }
}

// ---- utility kernels -------------------------------------------------------
__global__ void cvt_f32_bf16_kernel(const float* __restrict__ s,
                                    bf16_t* __restrict__ d, int n) {
  for (int i = blockIdx.x * blockDim.x + threadIdx.x; i < n;
       i += gridDim.x * blockDim.x)
    d[i] = (bf16_t)s[i];
}

// strided convert of x[:, t, :] (row stride SEQ*INP) -> bf16 [64,1024]
__global__ void cvt_xslice_kernel(const float* __restrict__ xt,
                                  bf16_t* __restrict__ d) {
  for (int i = blockIdx.x * blockDim.x + threadIdx.x; i < BATCH * INP;
       i += gridDim.x * blockDim.x) {
    const int row = i >> 10, col = i & (INP - 1);
    d[i] = (bf16_t)xt[(long)row * (SEQ * INP) + col];
  }
}

__global__ void zero_state_kernel(float* __restrict__ pf,
                                  bf16_t* __restrict__ pb, int n) {
  for (int i = blockIdx.x * blockDim.x + threadIdx.x; i < n;
       i += gridDim.x * blockDim.x) {
    pf[i] = 0.f;
    pb[i] = (bf16_t)0.f;
  }
}

// ---- host side -------------------------------------------------------------
extern "C" void kernel_launch(void* const* d_in, const int* in_sizes, int n_in,
                              void* d_out, int out_size, void* d_ws,
                              size_t ws_size, hipStream_t stream) {
  (void)in_sizes; (void)n_in; (void)out_size; (void)ws_size;

  const float* x = (const float*)d_in[0];
  const float *wih_f[4], *whh_f[4], *bih_f[4], *bhh_f[4], *lin_f[4],
      *blin_f[4], *bng_f[4], *bnb_f[4], *exp_f[4], *bexp_f[4];
  for (int L = 0; L < 4; ++L) {
    const int base = 1 + L * 10;
    wih_f[L]  = (const float*)d_in[base + 0];
    whh_f[L]  = (const float*)d_in[base + 1];
    bih_f[L]  = (const float*)d_in[base + 2];
    bhh_f[L]  = (const float*)d_in[base + 3];
    lin_f[L]  = (const float*)d_in[base + 4];
    blin_f[L] = (const float*)d_in[base + 5];
    bng_f[L]  = (const float*)d_in[base + 6];
    bnb_f[L]  = (const float*)d_in[base + 7];
    exp_f[L]  = (const float*)d_in[base + 8];
    bexp_f[L] = (const float*)d_in[base + 9];
  }
  const float* projw_f = (const float*)d_in[41];
  const float* projb_f = (const float*)d_in[42];

  // workspace partition (bf16 weights stay L2-resident across all 512 steps)
  char* p = (char*)d_ws;
  auto take = [&](size_t bytes) -> void* {
    void* r = (void*)p;
    p += (bytes + 255) & ~(size_t)255;
    return r;
  };
  bf16_t *wihb[4], *whhb[4], *linb[4], *expb[4];
  for (int L = 0; L < 4; ++L) {
    wihb[L] = (bf16_t*)take((size_t)3 * HID * HID * 2);
    whhb[L] = (bf16_t*)take((size_t)3 * HID * HID * 2);
    linb[L] = (bf16_t*)take((size_t)HID * HID * 2);
    expb[L] = (bf16_t*)take((size_t)HID * NPOOL * 2);
  }
  bf16_t* projb = (bf16_t*)take((size_t)OUTD * HID * 2);
  float*  h     = (float*)take((size_t)BATCH * HID * 4);
  bf16_t* hbf   = (bf16_t*)take((size_t)BATCH * HID * 2);
  bf16_t* xbf   = (bf16_t*)take((size_t)BATCH * INP * 2);
  bf16_t* pool  = (bf16_t*)take((size_t)BATCH * NPOOL * 2);

  // one-time per call: weight down-conversion + state init
  for (int L = 0; L < 4; ++L) {
    cvt_f32_bf16_kernel<<<512, 256, 0, stream>>>(wih_f[L], wihb[L], 3 * HID * HID);
    cvt_f32_bf16_kernel<<<512, 256, 0, stream>>>(whh_f[L], whhb[L], 3 * HID * HID);
    cvt_f32_bf16_kernel<<<512, 256, 0, stream>>>(lin_f[L], linb[L], HID * HID);
    cvt_f32_bf16_kernel<<<256, 256, 0, stream>>>(exp_f[L], expb[L], HID * NPOOL);
  }
  cvt_f32_bf16_kernel<<<256, 256, 0, stream>>>(projw_f, projb, OUTD * HID);
  zero_state_kernel<<<64, 256, 0, stream>>>(h, hbf, BATCH * HID);

  float* out = (float*)d_out;
  for (int t = 0; t < SEQ; ++t) {
    cvt_xslice_kernel<<<64, 256, 0, stream>>>(x + (long)t * INP, xbf);
    gate_bn_pool_kernel<<<64, 128, 0, stream>>>(
        xbf, hbf, h, wihb[0], whhb[0], linb[0], bih_f[0], bhh_f[0], blin_f[0],
        bng_f[0], bnb_f[0], pool);
    expand_kernel<<<64, 128, 0, stream>>>(pool, expb[0], bexp_f[0], h, hbf);
    for (int L = 1; L < 4; ++L) {
      gate_bn_pool_kernel<<<64, 128, 0, stream>>>(
          hbf, hbf, h, wihb[L], whhb[L], linb[L], bih_f[L], bhh_f[L],
          blin_f[L], bng_f[L], bnb_f[L], pool);
      expand_kernel<<<64, 128, 0, stream>>>(pool, expb[L], bexp_f[L], h, hbf);
    }
    proj_kernel<<<32, 128, 0, stream>>>(hbf, projb, projb_f,
                                        out + (long)t * OUTD,
                                        (long)SEQ * OUTD);
  }
}
